// PointnetFPModule_61486751810146
// MI455X (gfx1250) — compile-verified, matrix operand order checked by reading, not run on hardware
//
#include <hip/hip_runtime.h>
#include <hip/hip_bf16.h>

// ---------------- problem constants (from reference setup_inputs) -----------
#define BATCH   8
#define NPTS    8192      // unknown points per batch
#define MPTS    2048      // known points per batch
#define C1      128       // unknow_feats channels
#define C2      256       // known_feats channels
#define CIN     384       // C1 + C2
#define HDIM    256       // MLP width
#define ROWS    (BATCH * NPTS)   // 65536 total points
#define BN_N    ((float)ROWS)

// ---------------- WMMA types ------------------------------------------------
typedef __attribute__((ext_vector_type(16))) __bf16 v16bf;
typedef __attribute__((ext_vector_type(8)))  float  v8f;

union Frag {
    v16bf v;
    uint4 u[2];   // 2 x 16 bytes = 16 bf16
};

__device__ __forceinline__ unsigned short f2bf(float f) {
    unsigned int u = __float_as_uint(f);
    unsigned int r = u + 0x7FFFu + ((u >> 16) & 1u);   // round-to-nearest-even
    return (unsigned short)(r >> 16);
}

// ---------------- kernel: f32 -> bf16 convert -------------------------------
__global__ void k_cvt_bf16(const float* __restrict__ src,
                           unsigned short* __restrict__ dst, int n) {
    int i = blockIdx.x * blockDim.x + threadIdx.x;
    if (i < n) dst[i] = f2bf(src[i]);
}

// ---------------- kernel: zero float buffer ---------------------------------
__global__ void k_zero(float* __restrict__ p, int n) {
    int i = blockIdx.x * blockDim.x + threadIdx.x;
    if (i < n) p[i] = 0.0f;
}

// ---------------- kernel: three_nn + interpolate + concat -> X1 bf16 --------
// grid: BATCH * (NPTS/256) blocks of 256 threads; one thread per unknown point
__global__ void k_prep(const float* __restrict__ unknown,       // (B,n,3)
                       const float* __restrict__ known,         // (B,m,3)
                       const float* __restrict__ unknow_feats,  // (B,n,C1)
                       const float* __restrict__ known_feats,   // (B,m,C2)
                       unsigned short* __restrict__ X1)         // (B*n, CIN) bf16
{
    __shared__ float kp[MPTS * 3];   // 24 KB of 320 KB LDS
    const int blocksPerBatch = NPTS / 256;
    const int b  = blockIdx.x / blocksPerBatch;
    const int pi = (blockIdx.x % blocksPerBatch) * 256 + threadIdx.x;

    const float* kb = known + (size_t)b * MPTS * 3;
    for (int t = threadIdx.x; t < MPTS * 3; t += 256) kp[t] = kb[t];
    __syncthreads();

    const size_t gp = (size_t)b * NPTS + pi;
    const float ux = unknown[gp * 3 + 0];
    const float uy = unknown[gp * 3 + 1];
    const float uz = unknown[gp * 3 + 2];

    float d0 = 3.4e38f, d1 = 3.4e38f, d2 = 3.4e38f;
    int   i0 = 0, i1 = 0, i2 = 0;
    for (int j = 0; j < MPTS; ++j) {
        float dx = ux - kp[3 * j + 0];
        float dy = uy - kp[3 * j + 1];
        float dz = uz - kp[3 * j + 2];
        float d  = dx * dx + dy * dy + dz * dz;
        if (d < d0)      { d2 = d1; i2 = i1; d1 = d0; i1 = i0; d0 = d; i0 = j; }
        else if (d < d1) { d2 = d1; i2 = i1; d1 = d;  i1 = j; }
        else if (d < d2) { d2 = d;  i2 = j; }
    }

    float r0 = 1.0f / (sqrtf(fmaxf(d0, 0.0f)) + 1e-10f);
    float r1 = 1.0f / (sqrtf(fmaxf(d1, 0.0f)) + 1e-10f);
    float r2 = 1.0f / (sqrtf(fmaxf(d2, 0.0f)) + 1e-10f);
    float rs = r0 + r1 + r2;
    float w0 = r0 / rs, w1 = r1 / rs, w2 = r2 / rs;

    const float* f0 = known_feats + ((size_t)b * MPTS + i0) * C2;
    const float* f1 = known_feats + ((size_t)b * MPTS + i1) * C2;
    const float* f2 = known_feats + ((size_t)b * MPTS + i2) * C2;
    unsigned short* xo = X1 + gp * CIN;

    for (int c = 0; c < C2; ++c)
        xo[c] = f2bf(w0 * f0[c] + w1 * f1[c] + w2 * f2[c]);
    const float* uf = unknow_feats + gp * C1;
    for (int c = 0; c < C1; ++c)
        xo[C2 + c] = f2bf(uf[c]);
}

// ---------------- kernel: bf16 WMMA GEMM  Y = A * W^T + bias ----------------
// A  : (R x K) bf16 row-major;  Wb : (N x K) bf16 row-major (so B(k,n)=Wb[n*K+k])
// Y  : (R x N) f32 row-major.
// One wave computes a 16x64 output slab: one shared A fragment feeds 4 WMMAs
// per k-step (4x A-traffic reuse; vmem:wmma = 2.5:1).
__global__ void k_gemm_bf16(const unsigned short* __restrict__ A,
                            const unsigned short* __restrict__ Wb,
                            const float* __restrict__ bias,
                            float* __restrict__ Y,
                            int R, int N, int K)
{
    const int wavesPerBlock = blockDim.x >> 5;
    const int waveId = blockIdx.x * wavesPerBlock + (threadIdx.x >> 5);
    const int slabsN = N >> 6;                 // 64-column slabs
    const int tileM  = waveId / slabsN;
    const int slabN  = waveId % slabsN;
    if (tileM * 16 >= R) return;

    const int lane = threadIdx.x & 31;
    const int half = lane >> 4;     // K-half selector per ISA 16-bit A layout
    const int lrow = lane & 15;

    const unsigned short* aPtr = A  + (size_t)(tileM * 16 + lrow) * K;
    const unsigned short* bPtr = Wb + (size_t)(slabN * 64 + lrow) * K;

    v8f acc0 = {}, acc1 = {}, acc2 = {}, acc3 = {};
    for (int k0 = 0; k0 < K; k0 += 32) {
        // lanes 0-15: K = {k0..k0+7, k0+16..k0+23}; lanes 16-31: +8 per chunk
        Frag a;
        a.u[0] = *(const uint4*)(aPtr + k0 +      half * 8);
        a.u[1] = *(const uint4*)(aPtr + k0 + 16 + half * 8);
        Frag b0, b1, b2, b3;
        b0.u[0] = *(const uint4*)(bPtr + 0 * 16 * K + k0 +      half * 8);
        b0.u[1] = *(const uint4*)(bPtr + 0 * 16 * K + k0 + 16 + half * 8);
        b1.u[0] = *(const uint4*)(bPtr + 1 * 16 * K + k0 +      half * 8);
        b1.u[1] = *(const uint4*)(bPtr + 1 * 16 * K + k0 + 16 + half * 8);
        b2.u[0] = *(const uint4*)(bPtr + 2 * 16 * K + k0 +      half * 8);
        b2.u[1] = *(const uint4*)(bPtr + 2 * 16 * K + k0 + 16 + half * 8);
        b3.u[0] = *(const uint4*)(bPtr + 3 * 16 * K + k0 +      half * 8);
        b3.u[1] = *(const uint4*)(bPtr + 3 * 16 * K + k0 + 16 + half * 8);
        acc0 = __builtin_amdgcn_wmma_f32_16x16x32_bf16(false, a.v, false, b0.v,
                                                       (short)0, acc0, false, false);
        acc1 = __builtin_amdgcn_wmma_f32_16x16x32_bf16(false, a.v, false, b1.v,
                                                       (short)0, acc1, false, false);
        acc2 = __builtin_amdgcn_wmma_f32_16x16x32_bf16(false, a.v, false, b2.v,
                                                       (short)0, acc2, false, false);
        acc3 = __builtin_amdgcn_wmma_f32_16x16x32_bf16(false, a.v, false, b3.v,
                                                       (short)0, acc3, false, false);
    }

    // C/D layout: lane<16 -> N=lane, M=0..7 ; lane>=16 -> N=lane-16, M=8..15
    const int rowBase = tileM * 16 + half * 8;
    v8f accs[4] = {acc0, acc1, acc2, acc3};
    for (int t = 0; t < 4; ++t) {
        const int col = slabN * 64 + t * 16 + lrow;
        const float bv = bias[col];
        for (int r = 0; r < 8; ++r)
            Y[(size_t)(rowBase + r) * N + col] = accs[t][r] + bv;
    }
}

// ---------------- kernel: per-channel sum / sumsq over rows -----------------
// grid = ROWS/256 blocks of 256 threads; thread owns one channel, block owns
// 256 consecutive rows (fully coalesced reads), one atomicAdd per channel.
__global__ void k_stats(const float* __restrict__ Y,
                        float* __restrict__ sum, float* __restrict__ sumsq)
{
    const int ch = threadIdx.x;                // 0..255
    const size_t row0 = (size_t)blockIdx.x * 256;
    float s = 0.0f, q = 0.0f;
    for (int r = 0; r < 256; ++r) {
        float v = Y[(row0 + r) * HDIM + ch];
        s += v; q += v * v;
    }
    atomicAdd(&sum[ch], s);
    atomicAdd(&sumsq[ch], q);
}

// ---------------- kernel: BN + ReLU, write bf16 (stage-1 activation) --------
__global__ void k_bn_relu_bf16(const float* __restrict__ Y,
                               const float* __restrict__ sum,
                               const float* __restrict__ sumsq,
                               const float* __restrict__ g,
                               const float* __restrict__ beta,
                               unsigned short* __restrict__ Xo)
{
    const size_t i = (size_t)blockIdx.x * blockDim.x + threadIdx.x;
    const int ch = (int)(i & (HDIM - 1));
    float mu  = sum[ch]   * (1.0f / BN_N);
    float var = sumsq[ch] * (1.0f / BN_N) - mu * mu;
    float inv = rsqrtf(var + 1e-5f);
    float v = (Y[i] - mu) * inv * g[ch] + beta[ch];
    Xo[i] = f2bf(fmaxf(v, 0.0f));
}

// ---------------- kernel: BN + ReLU, write f32 (final output) ---------------
__global__ void k_bn_relu_f32(const float* __restrict__ Y,
                              const float* __restrict__ sum,
                              const float* __restrict__ sumsq,
                              const float* __restrict__ g,
                              const float* __restrict__ beta,
                              float* __restrict__ out)
{
    const size_t i = (size_t)blockIdx.x * blockDim.x + threadIdx.x;
    const int ch = (int)(i & (HDIM - 1));
    float mu  = sum[ch]   * (1.0f / BN_N);
    float var = sumsq[ch] * (1.0f / BN_N) - mu * mu;
    float inv = rsqrtf(var + 1e-5f);
    float v = (Y[i] - mu) * inv * g[ch] + beta[ch];
    out[i] = fmaxf(v, 0.0f);
}

// ---------------- host launch ----------------------------------------------
extern "C" void kernel_launch(void* const* d_in, const int* in_sizes, int n_in,
                              void* d_out, int out_size, void* d_ws, size_t ws_size,
                              hipStream_t stream) {
    const float* unknown      = (const float*)d_in[0];
    const float* known        = (const float*)d_in[1];
    const float* unknow_feats = (const float*)d_in[2];
    const float* known_feats  = (const float*)d_in[3];
    const float* W1 = (const float*)d_in[4];
    const float* b1 = (const float*)d_in[5];
    const float* g1 = (const float*)d_in[6];
    const float* be1 = (const float*)d_in[7];
    const float* W2 = (const float*)d_in[8];
    const float* b2 = (const float*)d_in[9];
    const float* g2 = (const float*)d_in[10];
    const float* be2 = (const float*)d_in[11];

    // workspace layout (all offsets 256B aligned)
    char* ws = (char*)d_ws;
    const size_t szX1   = (size_t)ROWS * CIN * sizeof(unsigned short);   // 50331648
    const size_t szY    = (size_t)ROWS * HDIM * sizeof(float);           // 67108864
    const size_t szW1b  = (size_t)HDIM * CIN * sizeof(unsigned short);   // 196608
    const size_t szW2b  = (size_t)HDIM * HDIM * sizeof(unsigned short);  // 131072
    unsigned short* X1  = (unsigned short*)(ws);
    float*          Y   = (float*)(ws + szX1);
    unsigned short* W1b = (unsigned short*)(ws + szX1 + szY);
    unsigned short* W2b = (unsigned short*)(ws + szX1 + szY + szW1b);
    float*          st  = (float*)(ws + szX1 + szY + szW1b + szW2b);     // 512 floats
    unsigned short* X2  = X1;   // X1 dead after GEMM1 -> reuse for stage-2 acts

    // 1) weights -> bf16
    k_cvt_bf16<<<(HDIM * CIN + 255) / 256, 256, 0, stream>>>(W1, W1b, HDIM * CIN);
    k_cvt_bf16<<<(HDIM * HDIM + 255) / 256, 256, 0, stream>>>(W2, W2b, HDIM * HDIM);

    // 2) three_nn + interpolate + concat -> X1 bf16
    k_prep<<<BATCH * (NPTS / 256), 256, 0, stream>>>(
        unknown, known, unknow_feats, known_feats, X1);

    // 3) GEMM1 (WMMA bf16): Y = X1 * W1^T + b1    [65536 x 384] x [384 x 256]
    {
        int waves = (ROWS / 16) * (HDIM / 64);   // 16384 waves, 8 waves/block
        k_gemm_bf16<<<waves / 8, 256, 0, stream>>>(X1, W1b, b1, Y, ROWS, HDIM, CIN);
    }

    // 4) BN1 stats + apply + ReLU -> X2 bf16
    k_zero<<<2, 256, 0, stream>>>(st, 512);
    k_stats<<<ROWS / 256, 256, 0, stream>>>(Y, st, st + 256);
    k_bn_relu_bf16<<<(ROWS * HDIM) / 256, 256, 0, stream>>>(Y, st, st + 256, g1, be1, X2);

    // 5) GEMM2 (WMMA bf16): Y = X2 * W2^T + b2    [65536 x 256] x [256 x 256]
    {
        int waves = (ROWS / 16) * (HDIM / 64);
        k_gemm_bf16<<<waves / 8, 256, 0, stream>>>(X2, W2b, b2, Y, ROWS, HDIM, HDIM);
    }

    // 6) BN2 stats + apply + ReLU -> d_out f32
    k_zero<<<2, 256, 0, stream>>>(st, 512);
    k_stats<<<ROWS / 256, 256, 0, stream>>>(Y, st, st + 256);
    k_bn_relu_f32<<<(ROWS * HDIM) / 256, 256, 0, stream>>>(Y, st, st + 256, g2, be2,
                                                           (float*)d_out);
}
